// LSTM_52364241272974
// MI455X (gfx1250) — matrix operand order repaired and checked by reading, not compiled, not run on hardware
//
#include <hip/hip_runtime.h>
#include <math.h>

// Problem constants (match reference)
#define B_    64
#define T_    512
#define I_    512
#define H_    1024
#define O_    1024
#define K0_   (I_ + H_)   // 1536
#define K1_   (2 * H_)    // 2048
#define NWG   64
#define SLICE 16          // hidden columns per workgroup (H_/NWG)

typedef __attribute__((ext_vector_type(16))) __bf16        v16bf;
typedef __attribute__((ext_vector_type(8)))  float         v8f;
typedef __attribute__((ext_vector_type(4)))  unsigned int  u32x4;   // POD 128-bit

union FragU { v16bf v; u32x4 q[2]; };

// Force global (AS1) access paths so the backend emits global_load_b128 /
// global_store_b16 even for laundered pointers (round-2 regressed to flat_*).
typedef __attribute__((address_space(1))) const u32x4    gc_u32x4;
typedef __attribute__((address_space(1))) unsigned short g_ushort;

__device__ __forceinline__ u32x4 ldg_b128(const unsigned short* p) {
  return *(gc_u32x4*)(unsigned long long)p;
}
__device__ __forceinline__ void stg_b16(unsigned short* p, unsigned short v) {
  *(g_ushort*)(unsigned long long)p = v;
}

__device__ __forceinline__ unsigned short f2bf(float f) {
  unsigned u = __float_as_uint(f);
  unsigned r = (u + 0x7FFFu + ((u >> 16) & 1u)) >> 16;  // RNE
  return (unsigned short)r;
}

// ---- WMMA fragment loaders (gfx1250 16-bit layouts, wave32) -----------------
// A fragment 16x32 (MxK): lane L<16 holds row m0+L, K = {k..k+7} U {k+16..k+23};
// lanes 16-31 hold the same rows with K shifted by +8.
__device__ __forceinline__ v16bf load_a(const unsigned short* __restrict__ src,
                                        int ld, int row0, int k0, int lane) {
  int r  = row0 + (lane & 15);
  int kh = (lane >> 4) << 3;                 // 0 or 8
  const unsigned short* p = src + (size_t)r * ld + k0 + kh;
  FragU f;
  f.q[0] = ldg_b128(p);                      // K = k0+kh .. +7
  f.q[1] = ldg_b128(p + 16);                 // K = k0+16+kh .. +7
  return f.v;
}

// B fragment 32x16 (KxN) from transposed weights Wt[N][K] row-major:
// lanes 0-15: col N=n0+lane, K={k0..k0+7} U {k0+16..k0+23}; lanes 16-31: K shifted by +8.
__device__ __forceinline__ v16bf load_b(const unsigned short* __restrict__ wt,
                                        int ld, int n0, int k0, int lane) {
  int n  = n0 + (lane & 15);
  int ks = (lane >> 4) << 3;                 // 0 or 8
  const unsigned short* p = wt + (size_t)n * ld + k0 + ks;
  FragU f;
  f.q[0] = ldg_b128(p);
  f.q[1] = ldg_b128(p + 16);
  return f.v;
}

__device__ __forceinline__ v8f wmma_bf16(v16bf a, v16bf b, v8f c) {
  return __builtin_amdgcn_wmma_f32_16x16x32_bf16(false, a, false, b,
                                                 (short)0, c, false, false);
}

// D tile (16x16 f32): VGPR r holds M=r (lanes 0-15) / M=8+r (lanes 16-31); N=lane&15.
__device__ __forceinline__ void store_gates(float* __restrict__ gates, int mtile,
                                            int ch, v8f acc, int lane) {
  int n     = lane & 15;
  int mbase = mtile * 16 + ((lane >> 4) << 3);
#pragma unroll
  for (int r = 0; r < 8; ++r)
    gates[(ch * B_ + (mbase + r)) * SLICE + n] = acc[r];
}

__device__ __forceinline__ float sigmoidf_(float x) { return 1.0f / (1.0f + __expf(-x)); }

// ---- device-wide barrier (persistent kernel, generation counter) ------------
__device__ __forceinline__ void grid_sync(unsigned* __restrict__ cnt,
                                          unsigned* __restrict__ gen) {
  __threadfence();          // release this WG's h-slice stores to device scope
  __syncthreads();
  if (threadIdx.x == 0) {
    unsigned g = __hip_atomic_load(gen, __ATOMIC_RELAXED, __HIP_MEMORY_SCOPE_AGENT);
    unsigned a = __hip_atomic_fetch_add(cnt, 1u, __ATOMIC_ACQ_REL, __HIP_MEMORY_SCOPE_AGENT);
    if (a == NWG - 1) {
      __hip_atomic_store(cnt, 0u, __ATOMIC_RELAXED, __HIP_MEMORY_SCOPE_AGENT);
      __hip_atomic_fetch_add(gen, 1u, __ATOMIC_RELEASE, __HIP_MEMORY_SCOPE_AGENT);
    } else {
      while (__hip_atomic_load(gen, __ATOMIC_ACQUIRE, __HIP_MEMORY_SCOPE_AGENT) == g)
        __builtin_amdgcn_s_sleep(8);
    }
  }
  __syncthreads();
  __threadfence();          // acquire: drop stale lines before reading peers' h
}

__device__ __forceinline__ void lstm_elem(const float* __restrict__ gates,
                                          float* __restrict__ cs,
                                          const float* __restrict__ bias,
                                          unsigned short* __restrict__ hdst,
                                          int j0, int tid) {
  for (int e = tid; e < B_ * SLICE; e += 256) {
    int b = e >> 4, j = e & 15;              // SLICE == 16
    float f  = gates[(0 * B_ + b) * SLICE + j] + bias[0 * H_ + j0 + j];
    float i  = gates[(1 * B_ + b) * SLICE + j] + bias[1 * H_ + j0 + j];
    float cc = gates[(2 * B_ + b) * SLICE + j] + bias[2 * H_ + j0 + j];
    float o  = gates[(3 * B_ + b) * SLICE + j] + bias[3 * H_ + j0 + j];
    float c  = sigmoidf_(f) * cs[e] + sigmoidf_(i) * tanhf(cc);
    cs[e] = c;
    stg_b16(hdst + (size_t)b * H_ + j0 + j, f2bf(sigmoidf_(o) * tanhf(c)));
  }
}

// ---- prologue: fp32 -> bf16 conversions -------------------------------------
__global__ void transpose_to_bf16(const float* __restrict__ src,
                                  unsigned short* __restrict__ dst, int R, int C) {
  int idx = blockIdx.x * blockDim.x + threadIdx.x;
  if (idx >= R * C) return;
  int c = idx % C, r = idx / C;              // coalesced read of src[R][C]
  dst[(size_t)c * R + r] = f2bf(src[idx]);   // dst[C][R]
}

__global__ void repack_x_bf16(const float* __restrict__ x,
                              unsigned short* __restrict__ xbf) {
  int idx = blockIdx.x * blockDim.x + threadIdx.x;   // x[B][T][I] -> xbf[T][B][I]
  if (idx >= B_ * T_ * I_) return;
  int i = idx % I_; int rest = idx / I_; int t = rest % T_; int b = rest / T_;
  xbf[((size_t)t * B_ + b) * I_ + i] = f2bf(x[idx]);
}

// ---- persistent 2-layer LSTM + FC epilogue ----------------------------------
__global__ __launch_bounds__(256, 1)
void lstm_persistent(const unsigned short* __restrict__ xbf,    // [T][B][I]
                     const unsigned short* __restrict__ w0t_in, // [4H][K0]
                     const unsigned short* __restrict__ w1t_in, // [4H][K1]
                     const unsigned short* __restrict__ wfct,   // [O][H]
                     const float* __restrict__ b0,
                     const float* __restrict__ b1,
                     const float* __restrict__ bfc,
                     unsigned short* __restrict__ hbuf,         // [4][B][H] bf16
                     float* __restrict__ out,                   // [B][O]
                     unsigned* __restrict__ bar) {
  __shared__ float gates[4 * B_ * SLICE];   // 16 KB: [chunk][b][j]
  __shared__ float csl[2][B_ * SLICE];      // 8 KB: persistent cell-state slices

  const int wg   = blockIdx.x;
  const int tid  = threadIdx.x;
  const int lane = tid & 31;
  const int wave = tid >> 5;
  const int j0   = wg * SLICE;
  const int mtile = wave >> 1;              // 0..3 (batch tile of 16)
  const int ch0   = (wave & 1) * 2;         // gate chunks {0,1} or {2,3}

  unsigned short* h0b[2] = { hbuf + 0 * (size_t)B_ * H_, hbuf + 1 * (size_t)B_ * H_ };
  unsigned short* h1b[2] = { hbuf + 2 * (size_t)B_ * H_, hbuf + 3 * (size_t)B_ * H_ };

  // zero this WG's h slices (all buffers) and cell state
  for (int e = tid; e < B_ * SLICE; e += 256) {
    int b = e >> 4, j = e & 15;
    csl[0][e] = 0.f; csl[1][e] = 0.f;
    size_t o = (size_t)b * H_ + j0 + j;
    stg_b16(h0b[0] + o, 0); stg_b16(h0b[1] + o, 0);
    stg_b16(h1b[0] + o, 0); stg_b16(h1b[1] + o, 0);
  }
  grid_sync(bar, bar + 1);

  for (int t = 0; t < T_; ++t) {
    const int pv = t & 1, cu = pv ^ 1;

    // Launder weight pointers each iteration: blocks LICM from hoisting the
    // (t-invariant) weight-fragment loads out of the time loop, which would
    // otherwise spill ~100 fragments to scratch (seen in round-1 disasm).
    // Loads through them still go to AS1 via ldg_b128.
    const unsigned short* w0t = w0t_in;
    const unsigned short* w1t = w1t_in;
    asm("" : "+s"(w0t), "+s"(w1t));

    // ---------------- layer 0: gates = [x_t | h0_prev] @ W0 ----------------
    {
      const unsigned short* xrow = xbf + (size_t)t * B_ * I_;
      // prefetch next step's x tile (cold HBM data) while we compute
      if (t + 1 < T_)
        __builtin_prefetch(xbf + (size_t)(t + 1) * B_ * I_ + (size_t)tid * 128, 0, 1);
      v8f acc0 = {}; v8f acc1 = {};
#pragma clang loop unroll_count(2)
      for (int k = 0; k < I_; k += 32) {                    // x part of K
        v16bf a  = load_a(xrow, I_, mtile * 16, k, lane);
        v16bf bA = load_b(w0t, K0_, (ch0 + 0) * H_ + j0, k, lane);
        v16bf bB = load_b(w0t, K0_, (ch0 + 1) * H_ + j0, k, lane);
        acc0 = wmma_bf16(a, bA, acc0);
        acc1 = wmma_bf16(a, bB, acc1);
      }
#pragma clang loop unroll_count(2)
      for (int k = 0; k < H_; k += 32) {                    // h0_prev part of K
        v16bf a  = load_a(h0b[pv], H_, mtile * 16, k, lane);
        v16bf bA = load_b(w0t, K0_, (ch0 + 0) * H_ + j0, I_ + k, lane);
        v16bf bB = load_b(w0t, K0_, (ch0 + 1) * H_ + j0, I_ + k, lane);
        acc0 = wmma_bf16(a, bA, acc0);
        acc1 = wmma_bf16(a, bB, acc1);
      }
      store_gates(gates, mtile, ch0 + 0, acc0, lane);
      store_gates(gates, mtile, ch0 + 1, acc1, lane);
    }
    __syncthreads();
    lstm_elem(gates, csl[0], b0, h0b[cu], j0, tid);
    grid_sync(bar, bar + 1);                // publish h0_t to all WGPs

    // ---------------- layer 1: gates = [h0_t | h1_prev] @ W1 ----------------
    {
      v8f acc0 = {}; v8f acc1 = {};
#pragma clang loop unroll_count(2)
      for (int k = 0; k < H_; k += 32) {                    // h0_t part of K
        v16bf a  = load_a(h0b[cu], H_, mtile * 16, k, lane);
        v16bf bA = load_b(w1t, K1_, (ch0 + 0) * H_ + j0, k, lane);
        v16bf bB = load_b(w1t, K1_, (ch0 + 1) * H_ + j0, k, lane);
        acc0 = wmma_bf16(a, bA, acc0);
        acc1 = wmma_bf16(a, bB, acc1);
      }
#pragma clang loop unroll_count(2)
      for (int k = 0; k < H_; k += 32) {                    // h1_prev part of K
        v16bf a  = load_a(h1b[pv], H_, mtile * 16, k, lane);
        v16bf bA = load_b(w1t, K1_, (ch0 + 0) * H_ + j0, H_ + k, lane);
        v16bf bB = load_b(w1t, K1_, (ch0 + 1) * H_ + j0, H_ + k, lane);
        acc0 = wmma_bf16(a, bA, acc0);
        acc1 = wmma_bf16(a, bB, acc1);
      }
      store_gates(gates, mtile, ch0 + 0, acc0, lane);
      store_gates(gates, mtile, ch0 + 1, acc1, lane);
    }
    __syncthreads();
    lstm_elem(gates, csl[1], b1, h1b[cu], j0, tid);
    grid_sync(bar, bar + 1);                // publish h1_t
  }

  // ---------------- epilogue: out = h1_final @ Wfc + bfc ----------------
  const unsigned short* hfin = h1b[((T_ - 1) & 1) ^ 1];
  if (wave < 4) {
    const int mt = wave;
    v8f acc = {};
#pragma clang loop unroll_count(2)
    for (int k = 0; k < H_; k += 32) {
      v16bf a = load_a(hfin, H_, mt * 16, k, lane);
      v16bf b = load_b(wfct, H_, j0, k, lane);
      acc = wmma_bf16(a, b, acc);
    }
    int n     = lane & 15;
    int mbase = mt * 16 + ((lane >> 4) << 3);
    float bias = bfc[j0 + n];
#pragma unroll
    for (int r = 0; r < 8; ++r)
      out[(size_t)(mbase + r) * O_ + j0 + n] = acc[r] + bias;
  }
}

// ---- host-side launch -------------------------------------------------------
extern "C" void kernel_launch(void* const* d_in, const int* in_sizes, int n_in,
                              void* d_out, int out_size, void* d_ws, size_t ws_size,
                              hipStream_t stream) {
  const float* x   = (const float*)d_in[0];
  const float* W0  = (const float*)d_in[1];
  const float* b0  = (const float*)d_in[2];
  const float* W1  = (const float*)d_in[3];
  const float* b1  = (const float*)d_in[4];
  const float* Wfc = (const float*)d_in[5];
  const float* bfc = (const float*)d_in[6];

  char* ws = (char*)d_ws;
  size_t off = 0;
  auto alloc = [&](size_t bytes) {
    char* p = ws + off;
    off += (bytes + 255) & ~(size_t)255;
    return p;
  };
  unsigned*       bar  = (unsigned*)alloc(256);
  unsigned short* xbf  = (unsigned short*)alloc((size_t)T_ * B_ * I_ * 2);
  unsigned short* w0t  = (unsigned short*)alloc((size_t)4 * H_ * K0_ * 2);
  unsigned short* w1t  = (unsigned short*)alloc((size_t)4 * H_ * K1_ * 2);
  unsigned short* wfct = (unsigned short*)alloc((size_t)O_ * H_ * 2);
  unsigned short* hbuf = (unsigned short*)alloc((size_t)4 * B_ * H_ * 2);
  (void)ws_size; (void)in_sizes; (void)n_in; (void)out_size;

  (void)hipMemsetAsync(bar, 0, 256, stream);   // barrier cnt/gen must start at 0

  int nt;
  nt = B_ * T_ * I_;  repack_x_bf16<<<(nt + 255) / 256, 256, 0, stream>>>(x, xbf);
  nt = K0_ * 4 * H_;  transpose_to_bf16<<<(nt + 255) / 256, 256, 0, stream>>>(W0, w0t, K0_, 4 * H_);
  nt = K1_ * 4 * H_;  transpose_to_bf16<<<(nt + 255) / 256, 256, 0, stream>>>(W1, w1t, K1_, 4 * H_);
  nt = H_ * O_;       transpose_to_bf16<<<(nt + 255) / 256, 256, 0, stream>>>(Wfc, wfct, H_, O_);

  lstm_persistent<<<NWG, 256, 0, stream>>>(xbf, w0t, w1t, wfct, b0, b1, bfc,
                                           hbuf, (float*)d_out, bar);
}
